// SetGNN_2594160246970
// MI455X (gfx1250) — matrix operand order repaired and checked by reading, not compiled, not run on hardware
//
#include <hip/hip_runtime.h>

#define DF 128
#define NN 100000
#define EE 20000
#define II 1000000

typedef __attribute__((ext_vector_type(16))) __bf16 v16bf;
typedef __attribute__((ext_vector_type(8)))  __bf16 v8bf;
typedef __attribute__((ext_vector_type(8)))  float  v8f;

// ---------------------------------------------------------------- utilities

__global__ void zero_f32(float* __restrict__ p, long n) {
  long i = (long)blockIdx.x * blockDim.x + threadIdx.x;
  if (i < n) p[i] = 0.0f;
}

__global__ void cvt_bf16(const float* __restrict__ s, __bf16* __restrict__ d, int n) {
  int i = blockIdx.x * blockDim.x + threadIdx.x;
  if (i < n) d[i] = (__bf16)s[i];
}

// one wave (32 lanes) per incidence: gather h[src] row, scale by norm,
// atomically accumulate into tot[dst] row; lane0 bumps the count.
__global__ void scatter_accum(const float* __restrict__ h,
                              const int* __restrict__ sidx,
                              const int* __restrict__ didx,
                              const float* __restrict__ nrm,
                              float* __restrict__ tot,
                              float* __restrict__ cnt, int nI) {
  int g = blockIdx.x * blockDim.x + threadIdx.x;
  int i = g >> 5;
  int lane = g & 31;
  if (i >= nI) return;
  int s = sidx[i], d = didx[i];
  float w = nrm[i];
  const float4* hp = (const float4*)(h + (size_t)s * DF);
  float4 v = hp[lane];
  float* tp = tot + (size_t)d * DF + lane * 4;
  atomicAdd(tp + 0, w * v.x);
  atomicAdd(tp + 1, w * v.y);
  atomicAdd(tp + 2, w * v.z);
  atomicAdd(tp + 3, w * v.w);
  if (lane == 0) atomicAdd(cnt + d, 1.0f);
}

__global__ void seg_mean(const float* __restrict__ tot, const float* __restrict__ cnt,
                         float* __restrict__ out, int rows) {
  int g = blockIdx.x * blockDim.x + threadIdx.x;
  int r = g >> 7;
  int c = g & 127;
  if (r >= rows) return;
  float cc = fmaxf(cnt[r], 1.0f);
  out[(size_t)r * DF + c] = tot[(size_t)r * DF + c] / cc;
}

// ---------------------------------------------------------------- fused MLP
// Y = act( act2( LN?(X) @ W0 + b0 ) @ W1 + b1 )
// act2 = relu(LN(.)) if LN else relu(.); act = relu if relu_out.
// Block: 256 threads = 8 waves, 128 rows per block, 16 rows per wave.
// A-frag (16-bit 16x32): lane L, elem e -> M=L%16, K = (e<8?e:e+8) + 8*(L>=16)
//   i.e. two contiguous 8-element runs: [kb, kb+8) and [kb+16, kb+24), kb = kt*32+8*(L>=16)
// B-frag (16-bit 32x16): lane L, elem e -> N=L%16, K = (e<8?e:e+8) + 8*(L>=16)
// C/D (f32 16x16): lane L, vgpr v -> M = v + 8*(L>=16), N = L%16

template <bool LN, int NTOUT>
__global__ __launch_bounds__(256, 1)
void mlp_fused(const float* __restrict__ X, int M,
               const __bf16* __restrict__ W0, const float* __restrict__ b0,
               const float* __restrict__ g0, const float* __restrict__ be0,
               const float* __restrict__ g1, const float* __restrict__ be1,
               const __bf16* __restrict__ W1, const float* __restrict__ b1,
               float* __restrict__ Y, int ncols, int relu_out)
{
  __shared__ __align__(32) __bf16 smem[2 * 16384];   // 64 KB
  __bf16* sW0 = smem;              // [kt][nt][lane][e], reused as sH after GEMM1
  __bf16* sW1 = smem + 16384;
  __bf16* sH  = smem;              // [wave][16][128]

  const int tid  = threadIdx.x;
  const int lane = tid & 31;
  const int wave = tid >> 5;
  const int half = lane >> 4;
  const int lr   = lane & 15;

  // ---- stage weights: vector row-chunk loads, scattered b16 stores into frag layout
  for (int o = tid; o < 2048; o += 256) {
    int k  = o >> 4;              // 0..127
    int n0 = (o & 15) * 8;        // 0,8,...,120
    int kt = k >> 5, kl = k & 31;
    int base = ((kt * 8 + (n0 >> 4)) * 32 + ((kl & 8) << 1) + (n0 & 15)) * 16 + ((kl & 7) | ((kl >> 1) & 8));
    v8bf w0 = *(const v8bf*)(W0 + k * DF + n0);
#pragma unroll
    for (int j = 0; j < 8; ++j) sW0[base + j * 16] = w0[j];
    v8bf w1;
    if (n0 < ncols) w1 = *(const v8bf*)(W1 + k * ncols + n0);
    else { v8bf z = {}; w1 = z; }
#pragma unroll
    for (int j = 0; j < 8; ++j) sW1[base + j * 16] = w1[j];
  }
  __syncthreads();

  const int row0 = (blockIdx.x * 8 + wave) * 16;
  const int arow = row0 + lr;
  const int rowc = (arow < M) ? arow : (M - 1);     // clamp: loads always valid
  const float* xr = X + (size_t)rowc * DF;

  // ---- input layernorm stats: lanes (lr, lr+16) split row in 64-col halves
  float mu = 0.0f, rs = 1.0f;
  if (LN) {
    float s = 0.0f, sq = 0.0f;
    const float* xs = xr + half * 64;
#pragma unroll
    for (int c = 0; c < 64; c += 4) {
      float4 v = *(const float4*)(xs + c);
      s  += v.x + v.y + v.z + v.w;
      sq += v.x * v.x + v.y * v.y + v.z * v.z + v.w * v.w;
    }
    s  += __shfl_xor(s, 16, 32);
    sq += __shfl_xor(sq, 16, 32);
    mu = s * (1.0f / DF);
    float var = sq * (1.0f / DF) - mu * mu;
    rs = rsqrtf(var + 1e-5f);
  }

  // ---- GEMM1: hidden = act(X) @ W0  (K=128 -> 4 k-tiles of 32)
  v8f zz = {};
  v8f acc[8];
#pragma unroll
  for (int nt = 0; nt < 8; ++nt) acc[nt] = zz;

#pragma unroll
  for (int kt = 0; kt < 4; ++kt) {
    const int kb = kt * 32 + half * 8;
    float xv[16];
    *(float4*)&xv[0]  = *(const float4*)(xr + kb);
    *(float4*)&xv[4]  = *(const float4*)(xr + kb + 4);
    *(float4*)&xv[8]  = *(const float4*)(xr + kb + 16);
    *(float4*)&xv[12] = *(const float4*)(xr + kb + 20);
    if (LN) {
      float gv[16], bv[16];
      *(float4*)&gv[0]  = *(const float4*)(g0 + kb);
      *(float4*)&gv[4]  = *(const float4*)(g0 + kb + 4);
      *(float4*)&gv[8]  = *(const float4*)(g0 + kb + 16);
      *(float4*)&gv[12] = *(const float4*)(g0 + kb + 20);
      *(float4*)&bv[0]  = *(const float4*)(be0 + kb);
      *(float4*)&bv[4]  = *(const float4*)(be0 + kb + 4);
      *(float4*)&bv[8]  = *(const float4*)(be0 + kb + 16);
      *(float4*)&bv[12] = *(const float4*)(be0 + kb + 20);
#pragma unroll
      for (int e = 0; e < 16; ++e) xv[e] = (xv[e] - mu) * rs * gv[e] + bv[e];
    }
    v16bf a;
#pragma unroll
    for (int e = 0; e < 16; ++e) a[e] = (__bf16)xv[e];
#pragma unroll
    for (int nt = 0; nt < 8; ++nt) {
      const v16bf b = *(const v16bf*)(sW0 + ((kt * 8 + nt) * 32 + lane) * 16);
      acc[nt] = __builtin_amdgcn_wmma_f32_16x16x32_bf16(
          false, a, false, b, (short)0, acc[nt], false, false);
    }
  }

  __syncthreads();   // everyone done reading sW0 -> reuse region as sH

  // ---- bias + hidden LN + relu on C-layout accumulators
  float rsum[8], rsq2[8];
#pragma unroll
  for (int v = 0; v < 8; ++v) { rsum[v] = 0.0f; rsq2[v] = 0.0f; }
#pragma unroll
  for (int nt = 0; nt < 8; ++nt) {
    float bb = b0[nt * 16 + lr];
#pragma unroll
    for (int v = 0; v < 8; ++v) {
      float hv = acc[nt][v] + bb;
      acc[nt][v] = hv;
      rsum[v] += hv; rsq2[v] += hv * hv;
    }
  }
  if (LN) {
#pragma unroll
    for (int m = 1; m <= 8; m <<= 1)
#pragma unroll
      for (int v = 0; v < 8; ++v) {
        rsum[v] += __shfl_xor(rsum[v], m, 32);
        rsq2[v] += __shfl_xor(rsq2[v], m, 32);
      }
  }
#pragma unroll
  for (int nt = 0; nt < 8; ++nt) {
    int n = nt * 16 + lr;
    float gg = LN ? g1[n] : 1.0f;
    float b2 = LN ? be1[n] : 0.0f;
#pragma unroll
    for (int v = 0; v < 8; ++v) {
      float hv = acc[nt][v];
      if (LN) {
        float m2 = rsum[v] * (1.0f / DF);
        float var = rsq2[v] * (1.0f / DF) - m2 * m2;
        hv = (hv - m2) * rsqrtf(var + 1e-5f) * gg + b2;
      }
      hv = fmaxf(hv, 0.0f);
      sH[(wave * 16 + v + half * 8) * DF + n] = (__bf16)hv;
    }
  }
  asm volatile("s_wait_dscnt 0x0" ::: "memory");  // same-wave LDS store->load

  // ---- GEMM2: out = hidden @ W1 (A-frag = two 16B LDS runs per k-tile)
  const __bf16* hrow = sH + (wave * 16 + lr) * DF;
  v8f acc2[NTOUT];
#pragma unroll
  for (int nt = 0; nt < NTOUT; ++nt) acc2[nt] = zz;
#pragma unroll
  for (int kt = 0; kt < 4; ++kt) {
    const int kb = kt * 32 + half * 8;
    v8bf lo = *(const v8bf*)(hrow + kb);
    v8bf hi = *(const v8bf*)(hrow + kb + 16);
    v16bf a = __builtin_shufflevector(lo, hi, 0, 1, 2, 3, 4, 5, 6, 7,
                                      8, 9, 10, 11, 12, 13, 14, 15);
#pragma unroll
    for (int nt = 0; nt < NTOUT; ++nt) {
      const v16bf b = *(const v16bf*)(sW1 + ((kt * 8 + nt) * 32 + lane) * 16);
      acc2[nt] = __builtin_amdgcn_wmma_f32_16x16x32_bf16(
          false, a, false, b, (short)0, acc2[nt], false, false);
    }
  }

  // ---- epilogue: bias (+relu) and store fp32
#pragma unroll
  for (int nt = 0; nt < NTOUT; ++nt) {
    int n = nt * 16 + lr;
    if (n >= ncols) continue;
    float bb = b1[n];
#pragma unroll
    for (int v = 0; v < 8; ++v) {
      int row = row0 + v + half * 8;
      if (row >= M) continue;
      float o = acc2[nt][v] + bb;
      if (relu_out) o = fmaxf(o, 0.0f);
      Y[(size_t)row * ncols + n] = o;
    }
  }
}

// ---------------------------------------------------------------- host side

extern "C" void kernel_launch(void* const* d_in, const int* in_sizes, int n_in,
                              void* d_out, int out_size, void* d_ws, size_t ws_size,
                              hipStream_t stream) {
  (void)in_sizes; (void)n_in; (void)out_size; (void)ws_size;

  const float* x    = (const float*)d_in[0];
  const int*   ei   = (const int*)d_in[1];
  const float* nrm  = (const float*)d_in[2];
  const int* src = ei;        // node ids, [I]
  const int* dst = ei + II;   // edge ids, [I]

  // params flattened: 8 MLPs (v2e0.enc, v2e0.dec, e2v0.enc, e2v0.dec, v2e1..)
  // each: W0,b0,W1,b1,g0,be0,g1,be1 ; then clf: W0,b0,W1,b1
  auto P = [&](int j, int t) { return (const float*)d_in[3 + j * 8 + t]; };
  const float* clfp[4];
  for (int t = 0; t < 4; ++t) clfp[t] = (const float*)d_in[3 + 64 + t];

  char* ws = (char*)d_ws;
  size_t offH = (size_t)NN * DF * sizeof(float);            // xcur at 0
  size_t offT = offH + (size_t)NN * DF * sizeof(float);
  size_t offC = offT + (size_t)NN * DF * sizeof(float);
  size_t offW = offC + (size_t)NN * sizeof(float);
  float*  xcur = (float*)ws;
  float*  bufH = (float*)(ws + offH);
  float*  bufT = (float*)(ws + offT);
  float*  bufC = (float*)(ws + offC);
  __bf16* wbf  = (__bf16*)(ws + offW);

  // convert all weight matrices to bf16 (layer W0/W1 pairs, then clf)
  for (int j = 0; j < 8; ++j) {
    cvt_bf16<<<64, 256, 0, stream>>>(P(j, 0), wbf + (size_t)j * 32768, 16384);
    cvt_bf16<<<64, 256, 0, stream>>>(P(j, 2), wbf + (size_t)j * 32768 + 16384, 16384);
  }
  __bf16* clfW0 = wbf + 8 * 32768;
  __bf16* clfW1 = clfW0 + 16384;
  cvt_bf16<<<64, 256, 0, stream>>>(clfp[0], clfW0, 16384);
  cvt_bf16<<<20, 256, 0, stream>>>(clfp[2], clfW1, DF * 40);

  auto mlp128 = [&](const float* in, int M, int j, float* out) {
    int blocks = (M + 127) / 128;
    mlp_fused<true, 8><<<blocks, 256, 0, stream>>>(
        in, M, wbf + (size_t)j * 32768, P(j, 1), P(j, 4), P(j, 5), P(j, 6), P(j, 7),
        wbf + (size_t)j * 32768 + 16384, P(j, 3), out, 128, 1);
  };

  auto conv = [&](const float* in, int n_in_rows, int n_out_rows,
                  const int* sidx, const int* didx, int jenc, int jdec, float* out) {
    mlp128(in, n_in_rows, jenc, bufH);                         // enc + relu
    long ztot = (long)n_out_rows * DF;
    zero_f32<<<(unsigned)((ztot + 255) / 256), 256, 0, stream>>>(bufT, ztot);
    zero_f32<<<(unsigned)((n_out_rows + 255) / 256), 256, 0, stream>>>(bufC, n_out_rows);
    scatter_accum<<<(II * 32 + 255) / 256, 256, 0, stream>>>(
        bufH, sidx, didx, nrm, bufT, bufC, II);
    seg_mean<<<(unsigned)((ztot + 255) / 256), 256, 0, stream>>>(
        bufT, bufC, bufH, n_out_rows);                         // mean -> bufH
    mlp128(bufH, n_out_rows, jdec, out);                       // dec + relu
  };

  const float* cur = x;
  for (int l = 0; l < 2; ++l) {
    conv(cur, NN, EE, src, dst, 4 * l + 0, 4 * l + 1, xcur);   // v2e
    cur = xcur;
    conv(cur, EE, NN, dst, src, 4 * l + 2, 4 * l + 3, xcur);   // e2v
  }

  // classifier: 128 -> 128 -> 40, no LN, no relu on output
  mlp_fused<false, 3><<<(NN + 127) / 128, 256, 0, stream>>>(
      cur, NN, clfW0, clfp[1], nullptr, nullptr, nullptr, nullptr,
      clfW1, clfp[3], (float*)d_out, 40, 0);
}
